// Attention_27118423507150
// MI455X (gfx1250) — matrix operand order repaired and checked by reading, not compiled, not run on hardware
//
#include <hip/hip_runtime.h>

typedef _Float16 h8   __attribute__((ext_vector_type(8)));
typedef _Float16 v16h __attribute__((ext_vector_type(16)));
typedef float    v8f  __attribute__((ext_vector_type(8)));
typedef int      i4   __attribute__((vector_size(16)));   // int4, matches builtin param

#define WMMA_F32_F16(a, b, c) \
  __builtin_amdgcn_wmma_f32_16x16x32_f16(false, (a), false, (b), (short)0, (c), false, false)

#if __has_builtin(__builtin_amdgcn_global_load_async_to_lds_b128) && \
    __has_builtin(__builtin_amdgcn_s_wait_asynccnt)
#define HAVE_ASYNC_LDS 1
#else
#define HAVE_ASYNC_LDS 0
#endif

#define GLOBAL_AS __attribute__((address_space(1)))
#define SHARED_AS __attribute__((address_space(3)))

namespace {
constexpr int   kL     = 2048;   // sequence length
constexpr int   kC     = 512;    // channels (= K dim of both projections)
constexpr int   kHeads = 8;
constexpr int   kD     = 64;     // head dim
constexpr float kScale = 0.125f; // 64^-0.5
}

union FragU { struct { h8 lo, hi; } p; v16h v; };

// A-matrix fragment, 16x32 f16.  Source row-major [16][stride] (K contiguous).
// Lane l<16: row=l, K chunks {0..7, 16..23}; lane l>=16: row=l-16, K {8..15, 24..31}.
__device__ __forceinline__ v16h frag_a(const _Float16* base, int stride, int lane) {
  const int r = lane & 15;
  const int k = (lane < 16) ? 0 : 8;
  const _Float16* p = base + r * stride + k;
  FragU u;
  u.p.lo = *(const h8*)(p);
  u.p.hi = *(const h8*)(p + 16);
  return u.v;
}

// B-matrix fragment, 32x16 f16.  Source column-major: column c's K-run is
// contiguous at base + c*stride.  Lane holds column (lane&15); lanes 0-15 take
// K {0..7, 16..23}, lanes 16-31 take K {8..15, 24..31} (matches frag_a).
__device__ __forceinline__ v16h frag_b(const _Float16* base, int stride, int lane) {
  const int c = lane & 15;
  const int k = (lane < 16) ? 0 : 8;
  const _Float16* p = base + c * stride + k;
  FragU u;
  u.p.lo = *(const h8*)(p);
  u.p.hi = *(const h8*)(p + 16);
  return u.v;
}

__device__ __forceinline__ v8f vzero() {
  v8f z = {0.f, 0.f, 0.f, 0.f, 0.f, 0.f, 0.f, 0.f};
  return z;
}

// Copy 16 f16 (32 bytes) global -> LDS.  Async path uses the CDNA5
// GLOBAL_LOAD_ASYNC_TO_LDS_B128 (ASYNCcnt-tracked, bypasses VGPRs).
__device__ __forceinline__ void stage16(_Float16* dst, const _Float16* src) {
#if HAVE_ASYNC_LDS
  __builtin_amdgcn_global_load_async_to_lds_b128(
      (GLOBAL_AS i4*)(src), (SHARED_AS i4*)(dst), 0, 0);
  __builtin_amdgcn_global_load_async_to_lds_b128(
      (GLOBAL_AS i4*)(src + 8), (SHARED_AS i4*)(dst + 8), 0, 0);
#else
  *(h8*)dst       = *(const h8*)src;
  *(h8*)(dst + 8) = *(const h8*)(src + 8);
#endif
}

__device__ __forceinline__ void stage_fence() {
#if HAVE_ASYNC_LDS
  __builtin_amdgcn_s_wait_asynccnt(0);
#endif
  __syncthreads();
}

// ---------------------------------------------------------------------------
// Packing kernels: run once, move fp32->f16 conversion + transpose OUT of the
// GEMM mainloops so those loops are pure async-copy + ds_load_b128 + wmma.
// ---------------------------------------------------------------------------
__global__ __launch_bounds__(256) void cvt_f16_kernel(
    const float* __restrict__ in, _Float16* __restrict__ out, int n) {
  const int i = blockIdx.x * 256 + threadIdx.x;
  if (i < n) out[i] = (_Float16)in[i];
}

// x[4][512][2048] fp32  ->  xT[4][2048][512] f16  (so GEMM-B K-runs are contiguous)
__global__ __launch_bounds__(256) void pack_xT_kernel(
    const float* __restrict__ x, _Float16* __restrict__ xT) {
  const int id = blockIdx.x * 256 + threadIdx.x;   // 4 * 2^20 elements
  const int l  = id & (kL - 1);
  const int c  = (id >> 11) & (kC - 1);
  const int n  = id >> 20;
  xT[((size_t)n * kL + l) * kC + c] = (_Float16)x[id];
}

// ---------------------------------------------------------------------------
// Shared 128x128 GEMM mainloop: acc += A(f16,[M][512]) * B, where B is given
// K-major as BT(f16,[2048][512]).  256 threads = 8 waves (4x2 wave grid),
// each wave owns 32x64 = 2x4 WMMA accumulators.  K-step 32.
// Both tiles stage as contiguous 16B transfers (async on CDNA5).
// ---------------------------------------------------------------------------
__device__ __forceinline__ void gemm_tile_128x128(
    const _Float16* __restrict__ A, const _Float16* __restrict__ BT,
    int m0, int l0, int tid, _Float16* Asub, _Float16* Bsub, v8f acc[2][4]) {
  const int lane = tid & 31;
  const int wid  = tid >> 5;
  const int wm   = wid >> 1;   // 0..3
  const int wn   = wid & 1;    // 0..1

  for (int mt = 0; mt < 2; ++mt)
    for (int nt = 0; nt < 4; ++nt) acc[mt][nt] = vzero();

  const int row = tid >> 1;          // 0..127
  const int kg  = (tid & 1) * 16;    // 0 or 16

  for (int k0 = 0; k0 < kC; k0 += 32) {
    // Asub[row][k] from A row-major; Bsub[col][k] from BT row-major — raw copies.
    stage16(&Asub[row * 32 + kg], A  + (size_t)(m0 + row) * kC + k0 + kg);
    stage16(&Bsub[row * 32 + kg], BT + (size_t)(l0 + row) * kC + k0 + kg);
    stage_fence();
#pragma unroll
    for (int mt = 0; mt < 2; ++mt) {
      v16h af = frag_a(&Asub[(wm * 32 + mt * 16) * 32], 32, lane);
#pragma unroll
      for (int nt = 0; nt < 4; ++nt) {
        v16h bf = frag_b(&Bsub[(wn * 64 + nt * 16) * 32], 32, lane);
        acc[mt][nt] = WMMA_F32_F16(af, bf, acc[mt][nt]);
      }
    }
    __syncthreads();
  }
}

// ---------------------------------------------------------------------------
// Kernel 1: QKV projection.  qkv = w_qkv(1536x512) @ x[n](512x2048).
// Epilogue scatters into attention-friendly f16 layouts:
//   q[n][h][i][d] (pre-scaled), k[n][h][j][d], v[n][h][d][j].
// ---------------------------------------------------------------------------
__global__ __launch_bounds__(256) void qkv_proj_kernel(
    const _Float16* __restrict__ wqkvh, const _Float16* __restrict__ xT,
    _Float16* __restrict__ qh, _Float16* __restrict__ kh,
    _Float16* __restrict__ vh) {
  __shared__ alignas(16) _Float16 Asub[128 * 32];
  __shared__ alignas(16) _Float16 Bsub[128 * 32];
  const int tid = threadIdx.x, lane = tid & 31, wid = tid >> 5;
  const int wm = wid >> 1, wn = wid & 1;
  const int m0 = blockIdx.x * 128, l0 = blockIdx.y * 128, n = blockIdx.z;

  v8f acc[2][4];
  gemm_tile_128x128(wqkvh, xT + (size_t)n * kL * kC, m0, l0, tid, Asub, Bsub, acc);

#pragma unroll
  for (int mt = 0; mt < 2; ++mt) {
#pragma unroll
    for (int nt = 0; nt < 4; ++nt) {
#pragma unroll
      for (int r = 0; r < 8; ++r) {
        const int m = m0 + wm * 32 + mt * 16 + r + ((lane < 16) ? 0 : 8);
        const int l = l0 + wn * 64 + nt * 16 + (lane & 15);
        const int h = (m >> 6) & 7;
        const int d = m & 63;
        const float val = acc[mt][nt][r];
        const size_t nh = (size_t)n * kHeads + h;
        if (m < 512) {
          qh[(nh * kL + l) * kD + d] = (_Float16)(val * kScale);
        } else if (m < 1024) {
          kh[(nh * kL + l) * kD + d] = (_Float16)val;
        } else {
          vh[(nh * kD + d) * kL + l] = (_Float16)val;
        }
      }
    }
  }
}

// ---------------------------------------------------------------------------
// Kernel 2: flash attention.  Each wave owns 16 query rows; loops over keys
// in chunks of 32 with online softmax; O accumulated 16x64 in fp32.
// K/V fragments come from L2-resident panels (256 KB per (n,h)); prefetch
// the next panel chunk into cache while the current WMMAs run.
// ---------------------------------------------------------------------------
__global__ __launch_bounds__(256) void flash_attn_kernel(
    const _Float16* __restrict__ qh, const _Float16* __restrict__ kh,
    const _Float16* __restrict__ vh, _Float16* __restrict__ ohT) {
  __shared__ alignas(16) _Float16 Pld[8][16 * 32];  // per-wave P staging
  const int tid = threadIdx.x, lane = tid & 31, wid = tid >> 5;
  const int n = blockIdx.z, h = blockIdx.y;
  const int i0 = blockIdx.x * 128 + wid * 16;
  const size_t nh = (size_t)n * kHeads + h;
  const _Float16* qb = qh + (nh * kL + i0) * kD;  // [16][64] row-major
  const _Float16* kb = kh + nh * kL * kD;         // [j][64]
  const _Float16* vb = vh + nh * kD * kL;         // [d][2048]

  // Q fragments reused for the whole key loop: resident in VGPRs.
  const v16h qa0 = frag_a(qb, kD, lane);        // d = 0..31
  const v16h qa1 = frag_a(qb + 32, kD, lane);   // d = 32..63

  v8f o[4];
#pragma unroll
  for (int dt = 0; dt < 4; ++dt) o[dt] = vzero();
  float mrow[8], lrow[8];
#pragma unroll
  for (int r = 0; r < 8; ++r) { mrow[r] = -1e30f; lrow[r] = 0.f; }

  _Float16* pw = &Pld[wid][0];

  for (int j0 = 0; j0 < kL; j0 += 32) {
    if (j0 + 64 < kL) {  // warm GL2/WGP$ for the next-but-one key/value chunk
      __builtin_prefetch((const void*)(kb + (size_t)(j0 + 64) * kD + lane * 32), 0, 1);
      __builtin_prefetch((const void*)(vb + (size_t)(lane & 3) * kL * 16 + j0 + 64), 0, 1);
    }
    // S = (scaled Q)^T K for 16 rows x 32 keys: two 16x16 fp32 fragments.
    v8f s0 = vzero(), s1 = vzero();
    {
      v16h k0a = frag_b(kb + (size_t)j0 * kD, kD, lane);
      v16h k0b = frag_b(kb + (size_t)j0 * kD + 32, kD, lane);
      s0 = WMMA_F32_F16(qa0, k0a, s0);
      s0 = WMMA_F32_F16(qa1, k0b, s0);
      v16h k1a = frag_b(kb + (size_t)(j0 + 16) * kD, kD, lane);
      v16h k1b = frag_b(kb + (size_t)(j0 + 16) * kD + 32, kD, lane);
      s1 = WMMA_F32_F16(qa0, k1a, s1);
      s1 = WMMA_F32_F16(qa1, k1b, s1);
    }
    // Online softmax: xor<16 shuffles reduce within each 16-lane half,
    // matching the C-fragment row split (rows r / r+8).
#pragma unroll
    for (int r = 0; r < 8; ++r) {
      float x = fmaxf(s0[r], s1[r]);
      x = fmaxf(x, __shfl_xor(x, 1));
      x = fmaxf(x, __shfl_xor(x, 2));
      x = fmaxf(x, __shfl_xor(x, 4));
      x = fmaxf(x, __shfl_xor(x, 8));
      const float mnew  = fmaxf(mrow[r], x);
      const float alpha = __expf(mrow[r] - mnew);
      mrow[r] = mnew;
      const float p0 = __expf(s0[r] - mnew);
      const float p1 = __expf(s1[r] - mnew);
      s0[r] = p0; s1[r] = p1;
      float rs = p0 + p1;
      rs += __shfl_xor(rs, 1);
      rs += __shfl_xor(rs, 2);
      rs += __shfl_xor(rs, 4);
      rs += __shfl_xor(rs, 8);
      lrow[r] = lrow[r] * alpha + rs;
#pragma unroll
      for (int dt = 0; dt < 4; ++dt) o[dt][r] *= alpha;
    }
    // Convert P from C-fragment layout to A-fragment layout via per-wave LDS.
    {
      const int rbase = (lane < 16) ? 0 : 8;
      const int col   = lane & 15;
#pragma unroll
      for (int r = 0; r < 8; ++r) {
        pw[(rbase + r) * 32 + col]      = (_Float16)s0[r];
        pw[(rbase + r) * 32 + col + 16] = (_Float16)s1[r];
      }
    }
    const v16h pa = frag_a(pw, 32, lane);  // same-wave LDS ordering: no barrier
    // O += P * V^T  (B columns = d, K-run along j contiguous in v[d][j]).
#pragma unroll
    for (int dt = 0; dt < 4; ++dt) {
      v16h vf = frag_b(vb + (size_t)(dt * 16) * kL + j0, kL, lane);
      o[dt] = WMMA_F32_F16(pa, vf, o[dt]);
    }
  }
  // Normalize; store to ohT[n][i][h*64+d] so the output GEMM stages raw tiles.
#pragma unroll
  for (int dt = 0; dt < 4; ++dt) {
#pragma unroll
    for (int r = 0; r < 8; ++r) {
      const int i = i0 + r + ((lane < 16) ? 0 : 8);
      const int d = dt * 16 + (lane & 15);
      const float val = o[dt][r] / lrow[r];
      ohT[((size_t)n * kL + i) * kC + h * kD + d] = (_Float16)val;
    }
  }
}

// ---------------------------------------------------------------------------
// Kernel 3: output projection.  out[n] = w_out(512x512) @ O[n](512x2048) + b,
// with O supplied K-major as ohT[n][2048][512].
// ---------------------------------------------------------------------------
__global__ __launch_bounds__(256) void out_proj_kernel(
    const _Float16* __restrict__ wouth, const _Float16* __restrict__ ohT,
    const float* __restrict__ b_out, float* __restrict__ out) {
  __shared__ alignas(16) _Float16 Asub[128 * 32];
  __shared__ alignas(16) _Float16 Bsub[128 * 32];
  const int tid = threadIdx.x, lane = tid & 31, wid = tid >> 5;
  const int wm = wid >> 1, wn = wid & 1;
  const int m0 = blockIdx.x * 128, l0 = blockIdx.y * 128, n = blockIdx.z;

  v8f acc[2][4];
  gemm_tile_128x128(wouth, ohT + (size_t)n * kL * kC, m0, l0, tid, Asub, Bsub, acc);

#pragma unroll
  for (int mt = 0; mt < 2; ++mt) {
#pragma unroll
    for (int nt = 0; nt < 4; ++nt) {
#pragma unroll
      for (int r = 0; r < 8; ++r) {
        const int m = m0 + wm * 32 + mt * 16 + r + ((lane < 16) ? 0 : 8);
        const int l = l0 + wn * 64 + nt * 16 + (lane & 15);
        out[((size_t)n * kC + m) * kL + l] = acc[mt][nt][r] + b_out[m];
      }
    }
  }
}

extern "C" void kernel_launch(void* const* d_in, const int* in_sizes, int n_in,
                              void* d_out, int out_size, void* d_ws, size_t ws_size,
                              hipStream_t stream) {
  const float* x     = (const float*)d_in[0];  // [4][512][2048]
  const float* w_qkv = (const float*)d_in[1];  // [1536][512]
  const float* w_out = (const float*)d_in[2];  // [512][512]
  const float* b_out = (const float*)d_in[3];  // [512]
  float* out = (float*)d_out;                  // [4][512][2048]

  char* ws = (char*)d_ws;
  const size_t MiB = (size_t)1 << 20;
  _Float16* wqkvh = (_Float16*)(ws);             // 1.5 MiB  [1536][512]
  _Float16* wouth = (_Float16*)(ws + 2 * MiB);   // 0.5 MiB  [512][512]
  _Float16* xT    = (_Float16*)(ws + 3 * MiB);   // 8 MiB    [4][2048][512]
  _Float16* qh    = (_Float16*)(ws + 11 * MiB);  // 8 MiB    [n][h][i][d]
  _Float16* kh    = (_Float16*)(ws + 19 * MiB);  // 8 MiB    [n][h][j][d]
  _Float16* vh    = (_Float16*)(ws + 27 * MiB);  // 8 MiB    [n][h][d][j]
  _Float16* ohT   = (_Float16*)(ws + 35 * MiB);  // 8 MiB    [n][i][h*64+d]

  dim3 blk(256);
  cvt_f16_kernel<<<(1536 * 512) / 256, blk, 0, stream>>>(w_qkv, wqkvh, 1536 * 512);
  cvt_f16_kernel<<<(512 * 512) / 256, blk, 0, stream>>>(w_out, wouth, 512 * 512);
  pack_xT_kernel<<<(4 * kC * kL) / 256, blk, 0, stream>>>(x, xT);

  dim3 g_qkv(1536 / 128, kL / 128, 4);
  qkv_proj_kernel<<<g_qkv, blk, 0, stream>>>(wqkvh, xT, qh, kh, vh);

  dim3 g_fa(kL / 128, kHeads, 4);
  flash_attn_kernel<<<g_fa, blk, 0, stream>>>(qh, kh, vh, ohT);

  dim3 g_out(kC / 128, kL / 128, 4);
  out_proj_kernel<<<g_out, blk, 0, stream>>>(wouth, ohT, b_out, out);
}
